// CustomAttentionClassifier_43619687858748
// MI455X (gfx1250) — matrix operand order repaired and checked by reading, not compiled, not run on hardware
//
#include <hip/hip_runtime.h>
#include <hip/hip_bf16.h>

// Problem constants (from reference)
#define VSZ  30522
#define DMODEL 256
#define SLEN 512
#define BATCH 64
#define HIDN 128
#define NCLS 16

typedef __attribute__((ext_vector_type(16))) __bf16 v16bf;
typedef __attribute__((ext_vector_type(8)))  float  v8f;

union FragU { unsigned int u[8]; v16bf v; };

static __device__ __forceinline__ unsigned short f2bf(float f) {
    unsigned int u = __float_as_uint(f);
    // round-to-nearest-even truncation to bf16
    u += 0x7FFFu + ((u >> 16) & 1u);
    return (unsigned short)(u >> 16);
}

// A-fragment (16x32 bf16, MxK): lane holds row L%16; VGPR i -> K pairs
//   i<4 : k = 2i + 8*half ; i>=4 : k = 16 + 2(i-4) + 8*half
// rowbase points at (row m, k-base) of a row-major [M][K] bf16 matrix.
static __device__ __forceinline__ v16bf frag_A(const unsigned short* rowbase, int half) {
    FragU f;
    const int ko = 8 * half;
#pragma unroll
    for (int i = 0; i < 4; ++i) {
        f.u[i]     = *(const unsigned int*)(rowbase + ko + 2 * i);
        f.u[i + 4] = *(const unsigned int*)(rowbase + ko + 16 + 2 * i);
    }
    return f.v;
}

// B-fragment (32x16 bf16, KxN): lane holds column L%16; VGPR i -> K pairs
//   i<4 : k = 2i + 8*half ; i>=4 : k = 16 + 2(i-4) + 8*half
// rowbase points at (row n, k-base) of a row-major [N][K] bf16 matrix (i.e. B^T storage).
static __device__ __forceinline__ v16bf frag_B(const unsigned short* rowbase, int half) {
    FragU f;
    const int ko = 8 * half;
#pragma unroll
    for (int i = 0; i < 4; ++i) {
        f.u[i]     = *(const unsigned int*)(rowbase + ko + 2 * i);
        f.u[i + 4] = *(const unsigned int*)(rowbase + ko + 16 + 2 * i);
    }
    return f.v;
}

// ---------------------------------------------------------------------------
// Kernel 1: embedding gather + sinusoidal positional encoding, cast to bf16
// x_bf[b][s][d] = bf16( emb[ids[b,s]][d] + PE(s,d) )
// ---------------------------------------------------------------------------
__global__ void k_embed(const int* __restrict__ ids,
                        const float* __restrict__ emb,
                        unsigned short* __restrict__ xbf) {
    int idx = blockIdx.x * blockDim.x + threadIdx.x;   // over B*S*D = 8388608
    if (idx >= BATCH * SLEN * DMODEL) return;
    int d = idx & (DMODEL - 1);
    int rest = idx >> 8;           // DMODEL = 256
    int s = rest & (SLEN - 1);
    int bs = rest;                 // flattened (b*S + s)
    int tok = ids[bs];
    float v = emb[tok * DMODEL + d];
    const float coeff = -9.210340371976184f / (float)DMODEL;  // -ln(10000)/D
    float freq = __expf(coeff * (float)(d & ~1));
    float ang = (float)s * freq;
    float pe = (d & 1) ? __cosf(ang) : __sinf(ang);
    xbf[idx] = f2bf(v + pe);
}

// ---------------------------------------------------------------------------
// Kernel 2: transpose + cast the three projection weights to bf16 [N][K]
// ---------------------------------------------------------------------------
__global__ void k_wcast(const float* __restrict__ Wq,
                        const float* __restrict__ Wk,
                        const float* __restrict__ Wv,
                        unsigned short* __restrict__ Wqt,
                        unsigned short* __restrict__ Wkt,
                        unsigned short* __restrict__ Wvt) {
    int idx = blockIdx.x * blockDim.x + threadIdx.x;   // over 3*D*D
    if (idx >= 3 * DMODEL * DMODEL) return;
    int which = idx / (DMODEL * DMODEL);
    int r = idx - which * (DMODEL * DMODEL);
    int k = r >> 8;             // input-dim row of W
    int n = r & (DMODEL - 1);   // output-dim col of W
    const float* W = (which == 0) ? Wq : (which == 1) ? Wk : Wv;
    unsigned short* Wt = (which == 0) ? Wqt : (which == 1) ? Wkt : Wvt;
    Wt[n * DMODEL + k] = f2bf(W[k * DMODEL + n]);
}

// ---------------------------------------------------------------------------
// Kernel 3: zero the pooled accumulator (atomics target)
// ---------------------------------------------------------------------------
__global__ void k_zero(float* __restrict__ p, int n) {
    int idx = blockIdx.x * blockDim.x + threadIdx.x;
    if (idx < n) p[idx] = 0.0f;
}

// ---------------------------------------------------------------------------
// Kernel 4: QKV projection GEMM via WMMA bf16, strip-mined.
// One wave computes a full 16x256 output strip (16 N-tiles) of (X @ W + b),
// holding all 8 A-fragments (K = 256) in VGPRs for 16x A-reuse.
// Q, K stored row-major [B*S][D]; V stored transposed [B][D][S].
// ---------------------------------------------------------------------------
__global__ void k_qkv(const unsigned short* __restrict__ xbf,
                      const unsigned short* __restrict__ Wqt,
                      const unsigned short* __restrict__ Wkt,
                      const unsigned short* __restrict__ Wvt,
                      const float* __restrict__ bq,
                      const float* __restrict__ bk,
                      const float* __restrict__ bv,
                      unsigned short* __restrict__ Qb,
                      unsigned short* __restrict__ Kb,
                      unsigned short* __restrict__ Vtb) {
    const int lane  = threadIdx.x & 31;
    const int l16   = lane & 15;
    const int lh    = lane >> 4;
    const int gwave = blockIdx.x * (blockDim.x >> 5) + (threadIdx.x >> 5);

    const int stripsPerMat = BATCH * SLEN / 16;        // 2048 16-row strips
    int which = gwave / stripsPerMat;                  // 0=Q, 1=K, 2=V
    int mt    = gwave - which * stripsPerMat;
    int m0    = mt * 16;

    const unsigned short* Wt  = (which == 0) ? Wqt : (which == 1) ? Wkt : Wvt;
    const float*          bia = (which == 0) ? bq  : (which == 1) ? bk  : bv;

    // Hoist the 8 A-fragments for this strip (K = 256): 64 VGPRs, reused 16x.
    const unsigned short* arow = xbf + (size_t)(m0 + l16) * DMODEL;
    v16bf ax[DMODEL / 32];
#pragma unroll
    for (int kk = 0; kk < DMODEL / 32; ++kk)
        ax[kk] = frag_A(arow + kk * 32, lh);

    for (int nt = 0; nt < DMODEL / 16; ++nt) {
        const int n0 = nt * 16;
        const unsigned short* brow = Wt + (size_t)(n0 + l16) * DMODEL;
        v8f c = {};
#pragma unroll
        for (int kk = 0; kk < DMODEL / 32; ++kk) {      // 8 WMMAs per tile
            v16bf b = frag_B(brow + kk * 32, lh);
            c = __builtin_amdgcn_wmma_f32_16x16x32_bf16(false, ax[kk], false, b,
                                                        (short)0, c, false, false);
        }
        const int   col  = n0 + l16;
        const float bval = bia[col];
#pragma unroll
        for (int r = 0; r < 8; ++r) {
            int row = m0 + r + 8 * lh;                 // global row in [0, B*S)
            unsigned short o = f2bf(c[r] + bval);
            if (which == 2) {
                int b_ = row >> 9;                     // SLEN = 512
                int s_ = row & (SLEN - 1);
                Vtb[((size_t)b_ * DMODEL + col) * SLEN + s_] = o;   // V transposed
            } else if (which == 0) {
                Qb[(size_t)row * DMODEL + col] = o;
            } else {
                Kb[(size_t)row * DMODEL + col] = o;
            }
        }
    }
}

// ---------------------------------------------------------------------------
// Kernel 5: fused attention per (batch, 16-query-row block).
// scores = Q Kt / 16 -> LDS; softmax in LDS; context = P @ V; column sums
// atomically accumulated into pooledAcc[b][d].
// One wave (32 threads) per workgroup; 48 KB LDS. The [B,S,S] score tensor
// never touches HBM.
// ---------------------------------------------------------------------------
__global__ void __launch_bounds__(32)
k_attn(const unsigned short* __restrict__ Q,
       const unsigned short* __restrict__ K,
       const unsigned short* __restrict__ Vt,
       float* __restrict__ pooledAcc) {
    __shared__ float          sc[16 * SLEN];      // 32 KB score rows
    __shared__ unsigned short pbuf[16 * SLEN];    // 16 KB softmax probs (bf16)

    const int lane = threadIdx.x & 31;
    const int l16  = lane & 15;
    const int lh   = lane >> 4;
    const int b    = blockIdx.x >> 5;             // S/16 = 32 row-blocks per batch
    const int m0   = (blockIdx.x & 31) * 16;

    const unsigned short* Qrow = Q  + ((size_t)b * SLEN + m0 + l16) * DMODEL;
    const unsigned short* Kbat = K  + (size_t)b * SLEN * DMODEL;
    const unsigned short* Vbat = Vt + (size_t)b * DMODEL * SLEN;

    // Hoist the 8 Q A-fragments (K = 256) — reused for all 32 key tiles.
    v16bf aq[DMODEL / 32];
#pragma unroll
    for (int kk = 0; kk < DMODEL / 32; ++kk)
        aq[kk] = frag_A(Qrow + kk * 32, lh);

    // ---- scores: 32 key tiles x 8 WMMAs, scaled by 1/sqrt(D) = 1/16 ----
    for (int t = 0; t < SLEN / 16; ++t) {
        const unsigned short* krow = Kbat + (size_t)(t * 16 + l16) * DMODEL;
        v8f c = {};
#pragma unroll
        for (int kk = 0; kk < DMODEL / 32; ++kk) {
            v16bf bfr = frag_B(krow + kk * 32, lh);
            c = __builtin_amdgcn_wmma_f32_16x16x32_bf16(false, aq[kk], false, bfr,
                                                        (short)0, c, false, false);
        }
        const int col = t * 16 + l16;
#pragma unroll
        for (int r = 0; r < 8; ++r)
            sc[(r + 8 * lh) * SLEN + col] = c[r] * 0.0625f;
    }
    __syncthreads();

    // ---- softmax: lanes 0..15 each own one query row ----
    if (lane < 16) {
        float* row = sc + lane * SLEN;
        float mx = -3.402823e38f;
        for (int j = 0; j < SLEN; ++j) mx = fmaxf(mx, row[j]);
        float sum = 0.0f;
        for (int j = 0; j < SLEN; ++j) { float e = __expf(row[j] - mx); row[j] = e; sum += e; }
        float inv = 1.0f / sum;
        unsigned short* prow = pbuf + lane * SLEN;
        for (int j = 0; j < SLEN; ++j) prow[j] = f2bf(row[j] * inv);
    }
    __syncthreads();

    // ---- context: 16 d-tiles x 16 WMMAs (K = 512); pooled column sums ----
    const unsigned short* prow = pbuf + (size_t)l16 * SLEN;
    for (int n = 0; n < DMODEL / 16; ++n) {
        const unsigned short* vrow = Vbat + (size_t)(n * 16 + l16) * SLEN;
        v8f c = {};
#pragma unroll
        for (int kk = 0; kk < SLEN / 32; ++kk) {
            v16bf a   = frag_A(prow + kk * 32, lh);     // LDS probs (ds loads)
            v16bf bfr = frag_B(vrow + kk * 32, lh);     // Vt [d][t] row-major
            c = __builtin_amdgcn_wmma_f32_16x16x32_bf16(false, a, false, bfr,
                                                        (short)0, c, false, false);
        }
        float s8 = 0.0f;
#pragma unroll
        for (int r = 0; r < 8; ++r) s8 += c[r];         // sum over 8 query rows
        // both half-lanes hit the same address; atomic combines them
        atomicAdd(&pooledAcc[b * DMODEL + n * 16 + l16], s8);
    }
}

// ---------------------------------------------------------------------------
// Kernel 6: tiny classifier head. pooled = acc/S; h = relu(pooled@Wc1+bc1);
// logits = h@Wc2+bc2. One block, 64 threads (thread = batch row).
// ---------------------------------------------------------------------------
__global__ void k_classifier(const float* __restrict__ pooledAcc,
                             const float* __restrict__ Wc1,
                             const float* __restrict__ bc1,
                             const float* __restrict__ Wc2,
                             const float* __restrict__ bc2,
                             float* __restrict__ out) {
    __shared__ float hsh[BATCH * HIDN];           // 32 KB
    int bb = threadIdx.x;
    if (bb < BATCH) {
        const float invS = 1.0f / (float)SLEN;
        for (int j = 0; j < HIDN; ++j) {
            float acc = bc1[j];
            for (int d = 0; d < DMODEL; ++d)
                acc += pooledAcc[bb * DMODEL + d] * invS * Wc1[d * HIDN + j];
            hsh[bb * HIDN + j] = acc > 0.0f ? acc : 0.0f;
        }
        for (int c = 0; c < NCLS; ++c) {
            float acc = bc2[c];
            for (int j = 0; j < HIDN; ++j)
                acc += hsh[bb * HIDN + j] * Wc2[j * NCLS + c];
            out[bb * NCLS + c] = acc;
        }
    }
}

// ---------------------------------------------------------------------------
extern "C" void kernel_launch(void* const* d_in, const int* in_sizes, int n_in,
                              void* d_out, int out_size, void* d_ws, size_t ws_size,
                              hipStream_t stream) {
    (void)in_sizes; (void)n_in; (void)out_size; (void)ws_size;

    const int*   ids = (const int*)  d_in[0];
    const float* emb = (const float*)d_in[1];
    const float* Wq  = (const float*)d_in[2];
    const float* bq  = (const float*)d_in[3];
    const float* Wk  = (const float*)d_in[4];
    const float* bk  = (const float*)d_in[5];
    const float* Wv  = (const float*)d_in[6];
    const float* bv  = (const float*)d_in[7];
    const float* Wc1 = (const float*)d_in[8];
    const float* bc1 = (const float*)d_in[9];
    const float* Wc2 = (const float*)d_in[10];
    const float* bc2 = (const float*)d_in[11];
    float* out = (float*)d_out;

    // workspace carve-out (bytes)
    char* ws = (char*)d_ws;
    const size_t XBYTES = (size_t)BATCH * SLEN * DMODEL * 2;     // 16 MB each
    const size_t WBYTES = (size_t)DMODEL * DMODEL * 2;           // 128 KB each
    unsigned short* xbf = (unsigned short*)(ws);
    unsigned short* Wqt = (unsigned short*)(ws + XBYTES);
    unsigned short* Wkt = (unsigned short*)(ws + XBYTES + WBYTES);
    unsigned short* Wvt = (unsigned short*)(ws + XBYTES + 2 * WBYTES);
    unsigned short* Qb  = (unsigned short*)(ws + XBYTES + 3 * WBYTES);
    unsigned short* Kb  = (unsigned short*)(ws + 2 * XBYTES + 3 * WBYTES);
    unsigned short* Vtb = (unsigned short*)(ws + 3 * XBYTES + 3 * WBYTES);
    float* pooledAcc    = (float*)        (ws + 4 * XBYTES + 3 * WBYTES);

    // 1) embed + PE -> bf16
    {
        int n = BATCH * SLEN * DMODEL;
        k_embed<<<(n + 255) / 256, 256, 0, stream>>>(ids, emb, xbf);
    }
    // 2) weight transpose/cast
    {
        int n = 3 * DMODEL * DMODEL;
        k_wcast<<<(n + 255) / 256, 256, 0, stream>>>(Wq, Wk, Wv, Wqt, Wkt, Wvt);
    }
    // 3) zero pooled accumulator
    {
        int n = BATCH * DMODEL;
        k_zero<<<(n + 255) / 256, 256, 0, stream>>>(pooledAcc, n);
    }
    // 4) QKV projections: 3 * 2048 strips, one wave per 16x256 strip
    {
        int waves  = 3 * (BATCH * SLEN / 16);      // 6144
        int blocks = waves / 8;                    // 8 waves (256 thr) per block
        k_qkv<<<blocks, 256, 0, stream>>>(xbf, Wqt, Wkt, Wvt, bq, bk, bv, Qb, Kb, Vtb);
    }
    // 5) fused attention + pooled reduction: one wave per (b, 16-row block)
    {
        int blocks = BATCH * (SLEN / 16);
        k_attn<<<blocks, 32, 0, stream>>>(Qb, Kb, Vtb, pooledAcc);
    }
    // 6) classifier head
    k_classifier<<<1, BATCH, 0, stream>>>(pooledAcc, Wc1, bc1, Wc2, bc2, out);
}
